// CrossAttentionFusion_71545565216957
// MI455X (gfx1250) — matrix operand order repaired and checked by reading, not compiled, not run on hardware
//
#include <hip/hip_runtime.h>

// ---------------------------------------------------------------------------
// CrossAttentionFusion on MI455X (gfx1250).
// seq_len==1 => softmax == 1 => attention collapses to
//   attended = (kv @ wv^T + bv) @ w_out^T + b_out
// Fold: Wt[n][k] = sum_j w_out[n][j] * wv[j][k]   (bf16, tiny GEMM)
//       bias[n] = w_out[n] . bv + b_out[n]
// Main: out[m][n] = sum_k feat[m][k] * Wt[n][k] + bias[n]  (bf16 WMMA, f32 acc)
//
// Memory plan: grid = (N-blocks fastest) so the 8 WGs sharing an A block-row
// are temporally adjacent -> feat rows served once from HBM, reused via L2.
// Output written with non-temporal stores (write-once stream).
// B tiles fetched with global_load_async_to_lds_b128 (ASYNCcnt), A tiles
// staged via VGPRs (f32->bf16) and hidden under WMMA. Double-buffered LDS.
// ---------------------------------------------------------------------------

typedef __attribute__((ext_vector_type(16))) __bf16 v16bf;
typedef __attribute__((ext_vector_type(8)))  __bf16 v8bf;
typedef __attribute__((ext_vector_type(4)))  __bf16 v4bf;
typedef __attribute__((ext_vector_type(8)))  float  v8f;

#define D_DIM  1024
#define B_ROWS 65536
#define LROW   40   // padded LDS row (bf16 elems): 80 B, 16 B aligned

static __device__ __forceinline__ __bf16 f2bf(float f) {
  unsigned u = __builtin_bit_cast(unsigned, f);
  u += 0x7FFFu + ((u >> 16) & 1u);            // round-to-nearest-even
  unsigned short h = (unsigned short)(u >> 16);
  return __builtin_bit_cast(__bf16, h);
}

// CDNA5 async copy: 16 B global -> LDS per lane, tracked by ASYNCcnt.
static __device__ __forceinline__ void async_load_b128(const void* gptr, void* lptr) {
  unsigned loff = (unsigned)(unsigned long long)lptr;  // low dword = LDS offset
  asm volatile("global_load_async_to_lds_b128 %0, %1, off"
               :: "v"(loff), "v"(gptr) : "memory");
}
static __device__ __forceinline__ void wait_asynccnt0() {
  asm volatile("s_wait_asynccnt 0x0" ::: "memory");
}

// ---------------------------------------------------------------------------
// out[k][j] = bf16(in[j][k])   (1024x1024 transpose + convert)
// ---------------------------------------------------------------------------
__global__ __launch_bounds__(256) void transpose_convert(
    const float* __restrict__ in, __bf16* __restrict__ out) {
  __shared__ float tile[32][33];
  const int tx = threadIdx.x, ty = threadIdx.y;   // 32 x 8
  const int bx = blockIdx.x * 32, by = blockIdx.y * 32;
#pragma unroll
  for (int i = 0; i < 4; ++i)
    tile[ty + i * 8][tx] = in[(by + ty + i * 8) * D_DIM + bx + tx];
  __syncthreads();
#pragma unroll
  for (int i = 0; i < 4; ++i)
    out[(bx + ty + i * 8) * D_DIM + by + tx] = f2bf(tile[tx][ty + i * 8]);
}

// ---------------------------------------------------------------------------
// bias[n] = dot(w_out[n][:], bv) + b_out[n]   (one WG of 256 per output)
// ---------------------------------------------------------------------------
__global__ __launch_bounds__(256) void bias_fold(
    const float* __restrict__ w_out, const float* __restrict__ bv,
    const float* __restrict__ b_out, float* __restrict__ bias) {
  __shared__ float red[256];
  const int n = blockIdx.x, t = threadIdx.x;
  float s = 0.f;
  for (int j = t; j < D_DIM; j += 256) s += w_out[n * D_DIM + j] * bv[j];
  red[t] = s;
  __syncthreads();
  for (int w = 128; w > 0; w >>= 1) {
    if (t < w) red[t] += red[t + w];
    __syncthreads();
  }
  if (t == 0) bias[n] = red[0] + b_out[n];
}

// ---------------------------------------------------------------------------
// C[m][n] = sum_k A_f32[m][k] * B_bf16[n][k]  (+ bias[n])
// blockIdx.x = N-block (fast-varying -> L2 reuse of A), blockIdx.y = M-block.
// WG = 256 threads = 8 waves (4x2), WG tile 128x128, wave tile 32x64, BK=32.
// OUTBF=1 -> bf16 output (weight fold), OUTBF=0 -> f32 NT output (main).
// ---------------------------------------------------------------------------
template <int OUTBF>
__global__ __launch_bounds__(256) void gemm_bf16(
    const float* __restrict__ A, const __bf16* __restrict__ Bw,
    const float* __restrict__ bias, void* __restrict__ Cv,
    int K, int ldc) {
  __shared__ __align__(16) __bf16 As[2][128 * LROW];
  __shared__ __align__(16) __bf16 Bs[2][128 * LROW];

  const int tid   = threadIdx.x;
  const int lane  = tid & 31;
  const int wave  = tid >> 5;
  const int wm    = wave >> 1;          // 0..3 : wave row
  const int wn    = wave & 1;           // 0..1 : wave col
  const int lx    = lane & 15;
  const int khalf = lane >> 4;

  const int nBlock = blockIdx.x * 128;  // fast-varying: siblings share A rows
  const int mBlock = blockIdx.y * 128;

  const v8f vzero = {0.f, 0.f, 0.f, 0.f, 0.f, 0.f, 0.f, 0.f};
  v8f acc[2][4];
#pragma unroll
  for (int i = 0; i < 2; ++i)
#pragma unroll
    for (int j = 0; j < 4; ++j) acc[i][j] = vzero;

  // per-thread load coordinates
  const int arow = tid >> 3, ac4 = tid & 7;   // A: 32 rows/pass x 8x(float4)
  const int brow = tid >> 2, bc8 = tid & 3;   // B: 64 rows/pass x 4x(v8bf)

  auto stage_a = [&](int kb, float4 ar[4]) {
#pragma unroll
    for (int p = 0; p < 4; ++p)
      ar[p] = *(const float4*)&A[(size_t)(mBlock + p * 32 + arow) * K + kb + ac4 * 4];
  };
  auto store_a = [&](int buf, const float4 ar[4]) {
#pragma unroll
    for (int p = 0; p < 4; ++p) {
      v4bf pk = {f2bf(ar[p].x), f2bf(ar[p].y), f2bf(ar[p].z), f2bf(ar[p].w)};
      *(v4bf*)&As[buf][(p * 32 + arow) * LROW + ac4 * 4] = pk;
    }
  };
  auto issue_b = [&](int kb, int buf) {
#pragma unroll
    for (int p = 0; p < 2; ++p)
      async_load_b128(&Bw[(size_t)(nBlock + p * 64 + brow) * K + kb + bc8 * 8],
                      &Bs[buf][(p * 64 + brow) * LROW + bc8 * 8]);
  };
  auto compute = [&](int buf) {
    // A 16x32 frag: lanes 0-15 hold K0-7/K16-23; lanes 16-31 hold K8-15/K24-31
    v16bf afr[2];
#pragma unroll
    for (int i = 0; i < 2; ++i) {
      const __bf16* rp = &As[buf][(wm * 32 + i * 16 + lx) * LROW + khalf * 8];
      v8bf lo = *(const v8bf*)rp;
      v8bf hi = *(const v8bf*)(rp + 16);
      afr[i] = __builtin_shufflevector(lo, hi, 0, 1, 2, 3, 4, 5, 6, 7,
                                       8, 9, 10, 11, 12, 13, 14, 15);
    }
    // B 32x16 frag: lane = N, lanes 0-15 K0-7/K16-23; lanes 16-31 K8-15/K24-31
    v16bf bfr[4];
#pragma unroll
    for (int j = 0; j < 4; ++j) {
      const __bf16* rp = &Bs[buf][(wn * 64 + j * 16 + lx) * LROW + khalf * 8];
      v8bf lo = *(const v8bf*)rp;
      v8bf hi = *(const v8bf*)(rp + 16);
      bfr[j] = __builtin_shufflevector(lo, hi, 0, 1, 2, 3, 4, 5, 6, 7,
                                       8, 9, 10, 11, 12, 13, 14, 15);
    }
#pragma unroll
    for (int i = 0; i < 2; ++i)
#pragma unroll
      for (int j = 0; j < 4; ++j)
        acc[i][j] = __builtin_amdgcn_wmma_f32_16x16x32_bf16(
            false, afr[i], false, bfr[j], (short)0, acc[i][j], false, false);
  };

  // ---- prologue: tile 0 -> buffer 0 ----
  {
    float4 ar[4];
    stage_a(0, ar);
    issue_b(0, 0);
    store_a(0, ar);
    wait_asynccnt0();
    __syncthreads();
  }

  // ---- pipelined main loop: one barrier per K-step ----
  int cur = 0;
  for (int kb = 32; kb < K; kb += 32) {
    float4 ar[4];
    stage_a(kb, ar);        // global f32 loads in flight during WMMA below
    issue_b(kb, cur ^ 1);   // async B -> other buffer
    if (kb + 32 < K)
      __builtin_prefetch(&A[(size_t)(mBlock + arow) * K + kb + 32 + ac4 * 4], 0, 3);
    compute(cur);
    store_a(cur ^ 1, ar);   // waits for A loads here, after compute
    wait_asynccnt0();
    __syncthreads();
    cur ^= 1;
  }
  compute(cur);

  // ---- epilogue: C/D layout lane=N, VGPR r = M row (+8 for upper half) ----
#pragma unroll
  for (int i = 0; i < 2; ++i) {
#pragma unroll
    for (int j = 0; j < 4; ++j) {
      const int gRow0 = mBlock + wm * 32 + i * 16 + khalf * 8;
      const int gCol  = nBlock + wn * 64 + j * 16 + lx;
      const float bb  = bias ? bias[gCol] : 0.f;
#pragma unroll
      for (int r = 0; r < 8; ++r) {
        const float v = acc[i][j][r] + bb;
        if (OUTBF) {
          ((__bf16*)Cv)[(size_t)(gRow0 + r) * ldc + gCol] = f2bf(v);
        } else {
          // write-once 512 MB stream: non-temporal, keep L2 for A reuse
          __builtin_nontemporal_store(v, &((float*)Cv)[(size_t)(gRow0 + r) * ldc + gCol]);
        }
      }
    }
  }
}

// ---------------------------------------------------------------------------
extern "C" void kernel_launch(void* const* d_in, const int* in_sizes, int n_in,
                              void* d_out, int out_size, void* d_ws, size_t ws_size,
                              hipStream_t stream) {
  (void)in_sizes; (void)n_in; (void)out_size; (void)ws_size;
  const float* feat_a   = (const float*)d_in[0];
  const float* feat_b   = (const float*)d_in[1];
  const float* w_in_ab  = (const float*)d_in[2];
  const float* b_in_ab  = (const float*)d_in[3];
  const float* w_out_ab = (const float*)d_in[4];
  const float* b_out_ab = (const float*)d_in[5];
  const float* w_in_ba  = (const float*)d_in[6];
  const float* b_in_ba  = (const float*)d_in[7];
  const float* w_out_ba = (const float*)d_in[8];
  const float* b_out_ba = (const float*)d_in[9];
  float* out = (float*)d_out;

  const size_t WBYTES = (size_t)D_DIM * D_DIM * sizeof(__bf16);  // 2 MB
  char* ws = (char*)d_ws;
  __bf16* Wt_ab  = (__bf16*)(ws + 0 * WBYTES);
  __bf16* Wt_ba  = (__bf16*)(ws + 1 * WBYTES);
  __bf16* wvT_ab = (__bf16*)(ws + 2 * WBYTES);
  __bf16* wvT_ba = (__bf16*)(ws + 3 * WBYTES);
  float*  bias_a = (float*)(ws + 4 * WBYTES);
  float*  bias_b = bias_a + D_DIM;

  const float* wv_ab = w_in_ab + 2 * D_DIM * D_DIM;  // value projection rows
  const float* wv_ba = w_in_ba + 2 * D_DIM * D_DIM;
  const float* bv_ab = b_in_ab + 2 * D_DIM;
  const float* bv_ba = b_in_ba + 2 * D_DIM;

  // 1) wvT[k][j] = bf16(wv[j][k])
  dim3 ttb(32, 8);
  transpose_convert<<<dim3(32, 32), ttb, 0, stream>>>(wv_ab, wvT_ab);
  transpose_convert<<<dim3(32, 32), ttb, 0, stream>>>(wv_ba, wvT_ba);

  // 2) fold: Wt[n][k] = sum_j w_out[n][j] * wvT[k][j]   (bf16 out)
  gemm_bf16<1><<<dim3(8, 8), 256, 0, stream>>>(w_out_ab, wvT_ab, nullptr,
                                               (void*)Wt_ab, D_DIM, D_DIM);
  gemm_bf16<1><<<dim3(8, 8), 256, 0, stream>>>(w_out_ba, wvT_ba, nullptr,
                                               (void*)Wt_ba, D_DIM, D_DIM);

  // 3) bias fold
  bias_fold<<<D_DIM, 256, 0, stream>>>(w_out_ab, bv_ab, b_out_ab, bias_a);
  bias_fold<<<D_DIM, 256, 0, stream>>>(w_out_ba, bv_ba, b_out_ba, bias_b);

  // 4) main GEMMs: out[:, :D] = feat_b @ Wt_ab^T + bias_a ; out[:, D:] likewise
  //    grid.x = N-blocks (fast) so sibling WGs reuse A rows from L2.
  gemm_bf16<0><<<dim3(D_DIM / 128, B_ROWS / 128), 256, 0, stream>>>(
      feat_b, Wt_ab, bias_a, (void*)out, D_DIM, 2 * D_DIM);
  gemm_bf16<0><<<dim3(D_DIM / 128, B_ROWS / 128), 256, 0, stream>>>(
      feat_a, Wt_ba, bias_b, (void*)(out + D_DIM), D_DIM, 2 * D_DIM);
}
